// ProductVectorQuantizer_30339648979486
// MI455X (gfx1250) — matrix operand order repaired and checked by reading, not compiled, not run on hardware
//
#include <hip/hip_runtime.h>

typedef __attribute__((ext_vector_type(2)))  float  v2f;
typedef __attribute__((ext_vector_type(8)))  float  v8f;
typedef __attribute__((ext_vector_type(16))) __bf16 v16bf;
typedef __attribute__((ext_vector_type(4)))  __bf16 v4bf;

#define DDIM 64
#define KCODES 1024
#define NROWS (256 * 512)
#define ROWS_PER_BLOCK 128
#define CODES_PER_GROUP 128
#define EPITCH 72   /* bf16 elems/row: 144B pitch -> each ds_load_b128 issue hits all 64 banks exactly 2x */

/* d_out layout (floats): quantized_sg | indices | loss */
#define QOFF 0
#define IOFF (NROWS * DDIM)     /* 8388608 */
#define LOFF (IOFF + NROWS)     /* 8519680 */

__device__ __forceinline__ __bf16 bf16_hi(float x) { return (__bf16)x; }
__device__ __forceinline__ __bf16 bf16_lo(float x, __bf16 h) { return (__bf16)(x - (float)h); }

/* ---- Kernel 1: X·E^T via bf16x3 split-precision V_WMMA_F32_16X16X32_BF16 + running argmin ----
 * S = (xh+xl)·(eh+el) ~= xh·eh + xh·el + xl·eh  (ll term ~2^-16 rel, negligible for argmin)
 * 6 bf16 WMMAs per 16x16x64 tile vs 16 f32 WMMAs: 2.67x fewer matrix ops, f32-grade argmin. */
__global__ __launch_bounds__(256)
void vq_argmin_kernel(const float* __restrict__ X, const float* __restrict__ E,
                      float* __restrict__ outIdxF) {
    __shared__ __bf16 ldsEh[CODES_PER_GROUP * EPITCH];
    __shared__ __bf16 ldsEl[CODES_PER_GROUP * EPITCH];
    __shared__ float  ldsN[CODES_PER_GROUP];

    const int tid  = threadIdx.x;
    const int wave = tid >> 5;
    const int lane = tid & 31;
    const int la   = lane & 15;
    const int hi   = lane >> 4;
    const int rowBase = blockIdx.x * ROWS_PER_BLOCK;
    const int myRow   = rowBase + wave * 16 + la;

    /* Preload this wave's 16x64 A tile as hi/lo bf16 fragments; reused for all 1024 codes.
       16-bit A 16x32 layout: lane(la,hi) elems 0..15 -> K=k0+16hi+j (contiguous, matches B). */
    v16bf ah[2], al[2];
    {
        const float* xr = X + (size_t)myRow * DDIM;
        #pragma unroll
        for (int s = 0; s < 2; ++s) {
            #pragma unroll
            for (int p = 0; p < 2; ++p) {
                const float* src = xr + 32 * s + 16 * hi + 8 * p;
                #pragma unroll
                for (int j = 0; j < 8; ++j) {
                    float  x = src[j];
                    __bf16 h = bf16_hi(x);
                    ah[s][8 * p + j] = h;
                    al[s][8 * p + j] = bf16_lo(x, h);
                }
            }
        }
    }

    float mval[8];
    int   midx[8];
    #pragma unroll
    for (int v = 0; v < 8; ++v) { mval[v] = 3.4e38f; midx[v] = 0; }

    const float4* Ev4 = (const float4*)E;

    for (int g = 0; g < KCODES / CODES_PER_GROUP; ++g) {
        const int codeBase = g * CODES_PER_GROUP;
        __syncthreads();
        /* Stage 128 codes: read f32 coalesced, split to hi/lo bf16 in padded LDS. */
        for (int i = tid; i < CODES_PER_GROUP * (DDIM / 4); i += 256) {
            int r = i >> 4, q = i & 15;
            float4 val = Ev4[(size_t)(codeBase + r) * (DDIM / 4) + q];
            v4bf h4, l4;
            {
                __bf16 h;
                h = bf16_hi(val.x); h4[0] = h; l4[0] = bf16_lo(val.x, h);
                h = bf16_hi(val.y); h4[1] = h; l4[1] = bf16_lo(val.y, h);
                h = bf16_hi(val.z); h4[2] = h; l4[2] = bf16_lo(val.z, h);
                h = bf16_hi(val.w); h4[3] = h; l4[3] = bf16_lo(val.w, h);
            }
            *(v4bf*)&ldsEh[r * EPITCH + 4 * q] = h4;
            *(v4bf*)&ldsEl[r * EPITCH + 4 * q] = l4;
        }
        /* exact f32 code norms straight from global (tiny: 32KB/block, L2-resident) */
        if (tid < CODES_PER_GROUP) {
            const float* er = E + (size_t)(codeBase + tid) * DDIM;
            float s = 0.f;
            #pragma unroll
            for (int j = 0; j < DDIM; ++j) s += er[j] * er[j];
            ldsN[tid] = s;
        }
        __syncthreads();

        #pragma unroll
        for (int cc = 0; cc < CODES_PER_GROUP / 16; ++cc) {
            const float nrm = ldsN[cc * 16 + la];
            v8f acc = {};
            #pragma unroll
            for (int s = 0; s < 2; ++s) {
                /* 16-bit B 32x16 layout: lane(la,hi) holds E[col=cc*16+la][k0+16hi .. +15] */
                const int off = (cc * 16 + la) * EPITCH + 16 * hi + 32 * s;
                v16bf bh, bl;
                __builtin_memcpy(&bh, &ldsEh[off], 32);
                __builtin_memcpy(&bl, &ldsEl[off], 32);
                acc = __builtin_amdgcn_wmma_f32_16x16x32_bf16(false, ah[s], false, bh,
                                                              (short)0, acc, false, false);
                acc = __builtin_amdgcn_wmma_f32_16x16x32_bf16(false, ah[s], false, bl,
                                                              (short)0, acc, false, false);
                acc = __builtin_amdgcn_wmma_f32_16x16x32_bf16(false, al[s], false, bh,
                                                              (short)0, acc, false, false);
            }
            const int col = codeBase + cc * 16 + la;
            #pragma unroll
            for (int v = 0; v < 8; ++v) {
                /* dist = ||x||^2 + ||e||^2 - 2 x.e ; row-constant ||x||^2 dropped (argmin-invariant) */
                float d = fmaf(-2.f, acc[v], nrm);
                if (d < mval[v]) { mval[v] = d; midx[v] = col; }
            }
        }
    }

    /* C/D layout: VGPR v, lane l -> row M = v + 8*(l>>4), col N = l&15.
       Min-reduce across the 16 lanes of each half-wave; lowest index wins ties. */
    #pragma unroll
    for (int v = 0; v < 8; ++v) {
        float m = mval[v];
        int   i = midx[v];
        #pragma unroll
        for (int off = 1; off < 16; off <<= 1) {
            float om = __shfl_xor(m, off, 32);
            int   oi = __shfl_xor(i, off, 32);
            if (om < m || (om == m && oi < i)) { m = om; i = oi; }
        }
        if (la == 0) {
            int row = rowBase + wave * 16 + v + 8 * hi;
            outIdxF[row] = (float)i;   /* indices are exact integers <= 1023 in f32 */
        }
    }
}

/* ---------------- Kernel 2: gather codebook row, straight-through output, loss ------------- */
__global__ __launch_bounds__(256)
void vq_gather_kernel(const float* __restrict__ X, const float* __restrict__ E,
                      const float* __restrict__ idxF,
                      float* __restrict__ outQ, float* __restrict__ outLoss) {
    const int tid  = threadIdx.x;
    const int row  = blockIdx.x * 16 + (tid >> 4);
    const int part = tid & 15;
    const int k    = (int)idxF[row];             /* broadcast load across the 16 lanes */

    const float4* Xv = (const float4*)X;
    const float4* Ev = (const float4*)E;
    float4 x = Xv[(size_t)row * 16 + part];
    float4 e = Ev[(size_t)k * 16 + part];

    float d0 = e.x - x.x, d1 = e.y - x.y, d2 = e.z - x.z, d3 = e.w - x.w;
    /* straight-through value computed exactly as reference: x + (q - x) */
    float4 q;
    q.x = x.x + d0; q.y = x.y + d1; q.z = x.z + d2; q.w = x.w + d3;
    ((float4*)outQ)[(size_t)row * 16 + part] = q;

    float s = d0 * d0 + d1 * d1 + d2 * d2 + d3 * d3;
    #pragma unroll
    for (int off = 1; off < 16; off <<= 1) s += __shfl_xor(s, off, 32);
    if (part == 0) outLoss[row] = s + 0.25f * s;  /* q_latent + commitment_cost * e_latent */
}

extern "C" void kernel_launch(void* const* d_in, const int* in_sizes, int n_in,
                              void* d_out, int out_size, void* d_ws, size_t ws_size,
                              hipStream_t stream) {
    (void)in_sizes; (void)n_in; (void)out_size; (void)d_ws; (void)ws_size;
    const float* X = (const float*)d_in[0];   /* [131072, 64] */
    const float* E = (const float*)d_in[1];   /* [1024, 64]   */
    float* out = (float*)d_out;

    vq_argmin_kernel<<<NROWS / ROWS_PER_BLOCK, 256, 0, stream>>>(X, E, out + IOFF);
    vq_gather_kernel<<<NROWS / 16, 256, 0, stream>>>(X, E, out + IOFF, out + QOFF, out + LOFF);
}
